// AGNN_49289044689239
// MI455X (gfx1250) — matrix operand order repaired and checked by reading, not compiled, not run on hardware
//
#include <hip/hip_runtime.h>
#include <math.h>

// ---------------------------------------------------------------------------
// AGNN (2-layer) forward for MI455X (gfx1250, wave32, WMMA + TDM).
//   h1 = relu(x @ W1 + b1)                [N,128]
//   h2 = AGNNConv(h1, beta=1)             [N,128]
//   h3 = AGNNConv(h2, beta=beta2)         [N,128]
//   out = log_softmax(h3 @ W2 + b2)       [N,64]
// GEMMs: A-panels DMA'd to LDS via TENSOR_LOAD_TO_LDS (D# with LDS padding to
// de-conflict banks), fp32->bf16 convert on fragment read, f32-accum
// v_wmma_f32_16x16x32_bf16.  Scatter-softmax: uint-encoded atomicMax +
// f32 atomicAdd (h/hn are 51MB -> L2-resident on 192MB L2).
// ---------------------------------------------------------------------------

typedef __attribute__((ext_vector_type(16))) __bf16 v16bf;
typedef __attribute__((ext_vector_type(8)))  float  v8f;
typedef __attribute__((ext_vector_type(4)))  unsigned u32x4;
typedef __attribute__((ext_vector_type(8)))  int      i32x8;
typedef __attribute__((ext_vector_type(4)))  int      i32x4;

#define C_IN  512
#define C_HID 128
#define C_OUT 64

__device__ __forceinline__ float wave_sum(float v) {
#pragma unroll
  for (int m = 16; m > 0; m >>= 1) v += __shfl_xor(v, m, 32);
  return v;
}
__device__ __forceinline__ float wave_max(float v) {
#pragma unroll
  for (int m = 16; m > 0; m >>= 1) v = fmaxf(v, __shfl_xor(v, m, 32));
  return v;
}
// order-preserving float<->uint encoding so atomicMax(uint) == float max
__device__ __forceinline__ unsigned f32_ord(float f) {
  unsigned u = __float_as_uint(f);
  return (u & 0x80000000u) ? ~u : (u | 0x80000000u);
}
__device__ __forceinline__ float ord_f32(unsigned u) {
  return __uint_as_float((u & 0x80000000u) ? (u & 0x7fffffffu) : ~u);
}

// ---------------------------------------------------------------------------
// Tensor Data Mover: DMA a 2-D fp32 tile (tile_d1 rows x tile_d0 elems,
// row stride stride_d0 elems) from global to LDS, optionally inserting
// pad_amount dwords into LDS every pad_interval dwords (bank de-conflict).
// D# layout per CDNA5 ISA (groups 2/3 zero => 2-D).
// NOTE: this toolchain exposes the 6-arg builtin:
//   (u32x4 g0, i32x8 g1, i32x4 g2, i32x4 g3, i32x8 unused, i32 cpol)
__device__ __forceinline__ void tdm_load_2d_f32(
    void* lds_ptr, const void* gptr,
    unsigned tile_d0, unsigned tile_d1,
    unsigned tensor_d0, unsigned tensor_d1,
    unsigned stride_d0,
    unsigned pad_interval_code, unsigned pad_amount_code, unsigned pad_en) {
  unsigned long long ga = (unsigned long long)gptr;
  unsigned lds_off = (unsigned)(unsigned long long)lds_ptr;  // low 32 = LDS byte addr
  u32x4 g0;
  g0[0] = 1u;                                                // count=1, user desc
  g0[1] = lds_off;                                           // lds_addr
  g0[2] = (unsigned)ga;                                      // global_addr[31:0]
  g0[3] = ((unsigned)(ga >> 32) & 0x01FFFFFFu) | (2u << 30); // addr[56:32] | type=2
  i32x8 g1;
  g1[0] = (int)((2u << 16) |                 // data_size = 4 bytes
                (pad_en << 20) |
                (pad_interval_code << 22) |
                (pad_amount_code << 25));
  g1[1] = (int)((tensor_d0 & 0xFFFFu) << 16);
  g1[2] = (int)(((tensor_d0 >> 16) & 0xFFFFu) | ((tensor_d1 & 0xFFFFu) << 16));
  g1[3] = (int)(((tensor_d1 >> 16) & 0xFFFFu) | (tile_d0 << 16));
  g1[4] = (int)(tile_d1 & 0xFFFFu);          // tile_dim1 | tile_dim2=0
  g1[5] = (int)stride_d0;                    // tensor_dim0_stride[31:0]
  g1[6] = 0;                                 // stride hi | dim1_stride lo
  g1[7] = 0;
  i32x4 z4 = {};
  i32x8 z8 = {};
  __builtin_amdgcn_tensor_load_to_lds(g0, g1, z4, z4, z8, 0);
}

// Build a 16x32 bf16 A-fragment from fp32 in LDS (row ptr rp at K base):
// vector elems 0..15 = K{0..15} relative to rp.
__device__ __forceinline__ v16bf a_frag_from_f32(const float* rp) {
  float4 f0 = *(const float4*)(rp + 0);
  float4 f1 = *(const float4*)(rp + 4);
  float4 f2 = *(const float4*)(rp + 8);
  float4 f3 = *(const float4*)(rp + 12);
  v16bf a;
  a[0] = (__bf16)f0.x;  a[1] = (__bf16)f0.y;  a[2] = (__bf16)f0.z;  a[3] = (__bf16)f0.w;
  a[4] = (__bf16)f1.x;  a[5] = (__bf16)f1.y;  a[6] = (__bf16)f1.z;  a[7] = (__bf16)f1.w;
  a[8] = (__bf16)f2.x;  a[9] = (__bf16)f2.y;  a[10] = (__bf16)f2.z; a[11] = (__bf16)f2.w;
  a[12] = (__bf16)f3.x; a[13] = (__bf16)f3.y; a[14] = (__bf16)f3.z; a[15] = (__bf16)f3.w;
  return a;
}

// ---------------------------------------------------------------------------
// W [K][Ncol] fp32  ->  wt [Ncol][K] bf16   (tiny, one-shot)
__global__ __launch_bounds__(256) void k_convert_wt(const float* __restrict__ w,
                                                    __bf16* __restrict__ wt,
                                                    int K, int Ncol) {
  int t = blockIdx.x * 256 + threadIdx.x;
  if (t >= K * Ncol) return;
  int k = t / Ncol, n = t % Ncol;
  wt[(size_t)n * K + k] = (__bf16)w[t];
}

__global__ __launch_bounds__(256) void k_zero(float4* __restrict__ p, long n4) {
  long i = (long)blockIdx.x * 256 + threadIdx.x;
  if (i < n4) p[i] = make_float4(0.f, 0.f, 0.f, 0.f);
}

// ---------------------------------------------------------------------------
// GEMM1: h = relu(x @ W1 + b1).  Block: 16 rows x 128 cols, 8 waves, one
// 16x16 tile each, K = 16 x 32.  A-slab (16x512 fp32, 32KB) arrives via one
// TDM DMA with 16B pad every 256 dwords -> row stride 2080B (bank spread, all
// fragment reads stay 16B aligned; pads land at k==256 => +16B iff kt>=8).
__global__ __launch_bounds__(256) void k_gemm1_relu(const float* __restrict__ x,
                                                    const __bf16* __restrict__ wt,  // [128][512]
                                                    const float* __restrict__ bias,
                                                    float* __restrict__ h, int N) {
  __shared__ __align__(16) unsigned char sX[16 * 2080];  // 16 rows, 2080B padded stride
  const int tid  = threadIdx.x;
  const int wave = tid >> 5, lane = tid & 31;
  const int hv   = lane >> 4, mn = lane & 15;
  const int r0   = blockIdx.x * 16;
  const int c0   = wave * 16;

  if (wave == 0) {
    tdm_load_2d_f32(sX, x + (size_t)r0 * C_IN,
                    /*tile*/ C_IN, 16, /*tensor*/ C_IN, 16, /*stride*/ C_IN,
                    /*pad: every 256dw (code 7) add 4dw (code 3)*/ 7, 3, 1);
    __builtin_amdgcn_s_wait_tensorcnt(0);
  }
  __syncthreads();

  v8f acc = {};
#pragma unroll
  for (int kt = 0; kt < 16; ++kt) {
    const float* rp = (const float*)(sX + mn * 2080 + (kt >= 8 ? 16 : 0))
                      + kt * 32 + hv * 16;
    v16bf a = a_frag_from_f32(rp);
    // B (32x16): lane n holds K{hv*16..hv*16+15}, contiguous in wt[n]
    v16bf b = *(const v16bf*)(wt + (size_t)(c0 + mn) * C_IN + kt * 32 + hv * 16);
    acc = __builtin_amdgcn_wmma_f32_16x16x32_bf16(false, a, false, b,
                                                  (short)0, acc, false, false);
  }

  const int col = c0 + mn;
  const float bb = bias[col];
#pragma unroll
  for (int i = 0; i < 8; ++i) {  // C: vgpr i -> row i + 8*half, col = lane&15
    float v = acc[i] + bb;
    h[(size_t)(r0 + i + hv * 8) * C_HID + col] = v > 0.f ? v : 0.f;
  }
}

// ---------------------------------------------------------------------------
// Row-normalize: hn = h / max(||h||, 1e-12).  One wave per row (128 feats).
__global__ __launch_bounds__(256) void k_normalize(const float* __restrict__ h,
                                                   float* __restrict__ hn, int N) {
  int row  = (blockIdx.x * 256 + threadIdx.x) >> 5;
  int lane = threadIdx.x & 31;
  if (row >= N) return;
  float4 f = *(const float4*)(h + (size_t)row * C_HID + lane * 4);
  float ss = wave_sum(f.x * f.x + f.y * f.y + f.z * f.z + f.w * f.w);
  float inv = 1.0f / fmaxf(sqrtf(ss), 1e-12f);
  float4 o = make_float4(f.x * inv, f.y * inv, f.z * inv, f.w * inv);
  *(float4*)(hn + (size_t)row * C_HID + lane * 4) = o;
}

// ---------------------------------------------------------------------------
// Edge logits + scatter-max.  One wave per edge; edges [E, E+N) are self-loops.
__global__ __launch_bounds__(256) void k_logit_max(const float* __restrict__ hn,
                                                   const int* __restrict__ srcI,
                                                   const int* __restrict__ dstI,
                                                   const float* __restrict__ beta_p,
                                                   float* __restrict__ logits,
                                                   unsigned* __restrict__ menc,
                                                   int E, int TOT) {
  int e    = (blockIdx.x * 256 + threadIdx.x) >> 5;
  int lane = threadIdx.x & 31;
  if (e >= TOT) return;
  int s_, d_;
  if (e < E) { s_ = srcI[e]; d_ = dstI[e]; } else { s_ = d_ = e - E; }
  float4 a = *(const float4*)(hn + (size_t)s_ * C_HID + lane * 4);
  float4 b = *(const float4*)(hn + (size_t)d_ * C_HID + lane * 4);
  float dot = wave_sum(a.x * b.x + a.y * b.y + a.z * b.z + a.w * b.w);
  float beta = beta_p ? *beta_p : 1.0f;
  float l = beta * dot;
  if (lane == 0) {
    logits[e] = l;
    atomicMax(menc + d_, f32_ord(l));
  }
}

// exp(logit - max[dst]) and scatter-sum of the exps.  One thread per edge.
__global__ __launch_bounds__(256) void k_expsum(float* __restrict__ logits,
                                                const unsigned* __restrict__ menc,
                                                float* __restrict__ ssum,
                                                const int* __restrict__ dstI,
                                                int E, int TOT) {
  int e = blockIdx.x * 256 + threadIdx.x;
  if (e >= TOT) return;
  int d_ = (e < E) ? dstI[e] : (e - E);
  float ev = expf(logits[e] - ord_f32(menc[d_]));
  logits[e] = ev;  // reuse buffer for exp values
  atomicAdd(ssum + d_, ev);
}

// out[dst] += (e / s[dst]) * h[src].  One wave per edge, 4 floats per lane.
__global__ __launch_bounds__(256) void k_aggregate(const float* __restrict__ h,
                                                   const float* __restrict__ evals,
                                                   const float* __restrict__ ssum,
                                                   const int* __restrict__ srcI,
                                                   const int* __restrict__ dstI,
                                                   float* __restrict__ out,
                                                   int E, int TOT) {
  int e    = (blockIdx.x * 256 + threadIdx.x) >> 5;
  int lane = threadIdx.x & 31;
  if (e >= TOT) return;
  int s_, d_;
  if (e < E) { s_ = srcI[e]; d_ = dstI[e]; } else { s_ = d_ = e - E; }
  float alpha = evals[e] / ssum[d_];
  float4 v = *(const float4*)(h + (size_t)s_ * C_HID + lane * 4);
  float* o = out + (size_t)d_ * C_HID + lane * 4;
  atomicAdd(o + 0, alpha * v.x);
  atomicAdd(o + 1, alpha * v.y);
  atomicAdd(o + 2, alpha * v.z);
  atomicAdd(o + 3, alpha * v.w);
}

// ---------------------------------------------------------------------------
// GEMM2 + log_softmax.  Block: 32 rows x 64 cols, 8 waves (2x4 tiles of 16x16),
// K=128.  A-slab (32x128 fp32, 16KB) via one TDM DMA, 16B pad per 128 dwords
// -> row stride 528B (lane m -> bank 4m, conflict-free); row-wise log-softmax
// through an LDS logits tile.
__global__ __launch_bounds__(256) void k_gemm2_lsm(const float* __restrict__ h,
                                                   const __bf16* __restrict__ wt2,  // [64][128]
                                                   const float* __restrict__ b2,
                                                   float* __restrict__ out, int N) {
  __shared__ __align__(16) unsigned char sX[32 * 528];
  __shared__ float sL[32][C_OUT];
  const int tid  = threadIdx.x;
  const int wave = tid >> 5, lane = tid & 31;
  const int hv   = lane >> 4, mn = lane & 15;
  const int r0   = blockIdx.x * 32;
  const int rw   = wave >> 2, cw = wave & 3;

  if (wave == 0) {
    tdm_load_2d_f32(sX, h + (size_t)r0 * C_HID,
                    /*tile*/ C_HID, 32, /*tensor*/ C_HID, 32, /*stride*/ C_HID,
                    /*pad: every 128dw (code 6) add 4dw (code 3)*/ 6, 3, 1);
    __builtin_amdgcn_s_wait_tensorcnt(0);
  }
  __syncthreads();

  v8f acc = {};
#pragma unroll
  for (int kt = 0; kt < 4; ++kt) {
    const float* rp = (const float*)(sX + (rw * 16 + mn) * 528) + kt * 32 + hv * 16;
    v16bf a = a_frag_from_f32(rp);
    v16bf b = *(const v16bf*)(wt2 + (size_t)(cw * 16 + mn) * C_HID + kt * 32 + hv * 16);
    acc = __builtin_amdgcn_wmma_f32_16x16x32_bf16(false, a, false, b,
                                                  (short)0, acc, false, false);
  }

  const int col = cw * 16 + mn;
  const float bb = b2[col];
#pragma unroll
  for (int i = 0; i < 8; ++i)
    sL[rw * 16 + i + hv * 8][col] = acc[i] + bb;
  __syncthreads();

  // log-softmax: wave w handles rows w, w+8, w+16, w+24; lane covers 2 cols
  for (int rr = wave; rr < 32; rr += 8) {
    float v0 = sL[rr][lane], v1 = sL[rr][lane + 32];
    float m = wave_max(fmaxf(v0, v1));
    float sum = wave_sum(expf(v0 - m) + expf(v1 - m));
    float lse = m + logf(sum);
    float* orow = out + (size_t)(r0 + rr) * C_OUT;
    orow[lane] = v0 - lse;
    orow[lane + 32] = v1 - lse;
  }
}

// ---------------------------------------------------------------------------
extern "C" void kernel_launch(void* const* d_in, const int* in_sizes, int n_in,
                              void* d_out, int out_size, void* d_ws, size_t ws_size,
                              hipStream_t stream) {
  const float* x     = (const float*)d_in[0];
  const int*   ei    = (const int*)d_in[1];   // [2,E] int32
  const float* W1    = (const float*)d_in[2];
  const float* b1    = (const float*)d_in[3];
  const float* beta2 = (const float*)d_in[4];
  const float* W2    = (const float*)d_in[5];
  const float* b2    = (const float*)d_in[6];
  float* out = (float*)d_out;

  const int N   = in_sizes[0] / C_IN;  // 100000
  const int E   = in_sizes[1] / 2;     // 1.6M
  const int TOT = E + N;               // + self-loops
  const int* srcI = ei;
  const int* dstI = ei + E;

  // workspace carve-up
  char* ws = (char*)d_ws;
  size_t off = 0;
  auto take = [&](size_t bytes) -> void* {
    void* p = ws + off;
    off = (off + bytes + 255) & ~(size_t)255;
    return p;
  };
  __bf16*   wt1    = (__bf16*)take((size_t)C_IN * C_HID * sizeof(__bf16));
  __bf16*   wt2    = (__bf16*)take((size_t)C_HID * C_OUT * sizeof(__bf16));
  float*    h1     = (float*)take((size_t)N * C_HID * sizeof(float));  // also h3
  float*    hn     = (float*)take((size_t)N * C_HID * sizeof(float));
  float*    h2     = (float*)take((size_t)N * C_HID * sizeof(float));
  float*    logits = (float*)take((size_t)TOT * sizeof(float));
  unsigned* menc   = (unsigned*)take((size_t)N * sizeof(unsigned));
  float*    ssum   = (float*)take((size_t)N * sizeof(float));

  const long nh4 = (long)N * C_HID / 4;
  const long nn4 = (long)(N + 3) / 4;
  auto blks = [](long n, int b) { return (unsigned)((n + b - 1) / b); };

  // weights: fp32 -> transposed bf16 (once per launch; deterministic)
  k_convert_wt<<<blks((long)C_IN * C_HID, 256), 256, 0, stream>>>(W1, wt1, C_IN, C_HID);
  k_convert_wt<<<blks((long)C_HID * C_OUT, 256), 256, 0, stream>>>(W2, wt2, C_HID, C_OUT);

  // h1 = relu(x @ W1 + b1)
  k_gemm1_relu<<<blks(N, 16), 256, 0, stream>>>(x, wt1, b1, h1, N);

  // ---- prop1 (beta = 1.0): h2 = AGNN(h1) ----
  k_zero<<<blks(nh4, 256), 256, 0, stream>>>((float4*)h2, nh4);
  k_zero<<<blks(nn4, 256), 256, 0, stream>>>((float4*)menc, nn4);
  k_zero<<<blks(nn4, 256), 256, 0, stream>>>((float4*)ssum, nn4);
  k_normalize<<<blks((long)N * 32, 256), 256, 0, stream>>>(h1, hn, N);
  k_logit_max<<<blks((long)TOT * 32, 256), 256, 0, stream>>>(hn, srcI, dstI, nullptr,
                                                             logits, menc, E, TOT);
  k_expsum<<<blks(TOT, 256), 256, 0, stream>>>(logits, menc, ssum, dstI, E, TOT);
  k_aggregate<<<blks((long)TOT * 32, 256), 256, 0, stream>>>(h1, logits, ssum, srcI,
                                                             dstI, h2, E, TOT);

  // ---- prop2 (beta = beta2): h3 = AGNN(h2), reusing h1's buffer ----
  float* h3 = h1;
  k_zero<<<blks(nh4, 256), 256, 0, stream>>>((float4*)h3, nh4);
  k_zero<<<blks(nn4, 256), 256, 0, stream>>>((float4*)menc, nn4);
  k_zero<<<blks(nn4, 256), 256, 0, stream>>>((float4*)ssum, nn4);
  k_normalize<<<blks((long)N * 32, 256), 256, 0, stream>>>(h2, hn, N);
  k_logit_max<<<blks((long)TOT * 32, 256), 256, 0, stream>>>(hn, srcI, dstI, beta2,
                                                             logits, menc, E, TOT);
  k_expsum<<<blks(TOT, 256), 256, 0, stream>>>(logits, menc, ssum, dstI, E, TOT);
  k_aggregate<<<blks((long)TOT * 32, 256), 256, 0, stream>>>(h2, logits, ssum, srcI,
                                                             dstI, h3, E, TOT);

  // out = log_softmax(h3 @ W2 + b2)
  k_gemm2_lsm<<<blks(N, 32), 256, 0, stream>>>(h3, wt2, b2, out, N);
}
